// PointCNNCls_79190607004308
// MI455X (gfx1250) — matrix operand order repaired and checked by reading, not compiled, not run on hardware
//
#include <hip/hip_runtime.h>

typedef __attribute__((ext_vector_type(16))) _Float16 v16h;
typedef __attribute__((ext_vector_type(8)))  _Float16 v8h;
typedef __attribute__((ext_vector_type(8)))  float    v8f;

#define BATCH 16
#define NPTS  2048
#define MAXSEL 73   // max K*D+1 = 12*6+1

static __device__ __forceinline__ int ceil32d(int x) { return (x + 31) & ~31; }

// ---------------------------------------------------------------------------
// (B,3,N) -> (B,N,3)
__global__ void pcnn_transpose_kernel(const float* __restrict__ x,
                                      float* __restrict__ pts, int Bv, int N) {
  int t = blockIdx.x * blockDim.x + threadIdx.x;
  int total = Bv * N * 3;
  if (t >= total) return;
  int d = t % 3; int n = (t / 3) % N; int b = t / (3 * N);
  pts[t] = x[((size_t)b * 3 + d) * N + n];
}

// ---------------------------------------------------------------------------
// Weight packing: W[Kd x Nd] (f32, row-major, ldw) -> Wt[Nd x kpad] (f16,
// transposed, zero-padded past Kd). Zero padding means the GEMM k-loop needs
// NO tail and NO predicates: A x 0 = 0 for any k >= Kd.
__global__ void pcnn_pack_w_kernel(const float* __restrict__ w, int ldw,
                                   _Float16* __restrict__ wt, int Kd, int Nd) {
  int kpad = ceil32d(Kd);
  int total = Nd * kpad;
  int t = blockIdx.x * blockDim.x + threadIdx.x;
  if (t >= total) return;
  int k = t % kpad; int n = t / kpad;
  wt[t] = (k < Kd) ? (_Float16)w[(size_t)k * ldw + n] : (_Float16)0.0f;
}

// xt0_W (K2,3,K) -> Wt[o][k*3+d]  (f16, transposed, padded)
__global__ void pcnn_pack_xt0_kernel(const float* __restrict__ w,
                                     _Float16* __restrict__ wt, int K) {
  int K2 = K * K, Kd = 3 * K, kpad = ceil32d(Kd);
  int total = K2 * kpad;
  int t = blockIdx.x * blockDim.x + threadIdx.x;
  if (t >= total) return;
  int r = t % kpad; int o = t / kpad;   // r = k*3 + d
  wt[t] = (r < Kd) ? (_Float16)w[((size_t)o * 3 + r % 3) * K + r / 3]
                   : (_Float16)0.0f;
}

// pw_W (Cout,Ccat,dm) -> Wt[o][c*dm+m]  (f16, transposed, padded)
__global__ void pcnn_pack_pw_kernel(const float* __restrict__ w,
                                    _Float16* __restrict__ wt,
                                    int Ccat, int dm, int Cout) {
  int Kd = Ccat * dm, kpad = ceil32d(Kd);
  int total = Cout * kpad;
  int t = blockIdx.x * blockDim.x + threadIdx.x;
  if (t >= total) return;
  int r = t % kpad; int o = t / kpad;   // r = c*dm + m
  wt[t] = (r < Kd) ? (_Float16)w[(size_t)o * Kd + r] : (_Float16)0.0f;
}

// ---------------------------------------------------------------------------
// GEMM: C[M,Nd] = opt_relu(A[M,Kd] * W[Kd,Nd] + bias) with W pre-packed as
// Wt[Nd x kpad] f16 (transposed, zero-padded). One wave -> 16x32 C tile,
// two v_wmma_f32_16x16x32_f16 per 32-deep k-step.
//  - M rows / N cols out of range: ADDRESS CLAMPED only (duplicated data lands
//    in C rows/cols the epilogue never stores).
//  - K padding: B is zero there, so A may read stale floats past its row end
//    (stays inside the workspace; harmless, multiplied by 0).
// => the k-loop has no predicates, no selects, no divergence; EXEC all-ones.
__global__ void pcnn_gemm_wmma_kernel(const float* __restrict__ A, int lda,
                                      const _Float16* __restrict__ Wt,
                                      const float* __restrict__ bias,
                                      float* __restrict__ C, int ldc,
                                      int M, int Kd, int Nd, int relu) {
  const int lane  = threadIdx.x;      // blockDim.x == 32 (one wave)
  const int hl    = lane >> 4;        // lane half
  const int l16   = lane & 15;
  const int tileN = blockIdx.x * 32;
  const int tileM = blockIdx.y * 16;
  const int kpad  = ceil32d(Kd);

  // A base: row clamp + lane-half K offset folded in (immediate offsets after)
  const int rowA = tileM + l16;
  const float* Ar = A + (size_t)(rowA < M ? rowA : M - 1) * lda + (hl ? 16 : 0);

  // B bases: column clamp + lane-half K offset folded in
  const int  colB0 = tileN + l16;
  const int  colB1 = colB0 + 16;
  const bool c0OK  = colB0 < Nd;
  const bool c1OK  = colB1 < Nd;
  const int  cB0   = c0OK ? colB0 : Nd - 1;
  const int  cB1   = c1OK ? colB1 : Nd - 1;
  const _Float16* Wc0 = Wt + (size_t)cB0 * kpad + (hl ? 16 : 0);
  const _Float16* Wc1 = Wt + (size_t)cB1 * kpad + (hl ? 16 : 0);

  v8f acc0, acc1;
#pragma unroll
  for (int r = 0; r < 8; ++r) { acc0[r] = 0.0f; acc1[r] = 0.0f; }

  for (int k0 = 0; k0 < kpad; k0 += 32) {
    __builtin_prefetch(&Ar[k0 + 32], 0, 1);         // global_prefetch_b8
    v16h a;
#pragma unroll
    for (int e = 0; e < 8; ++e) {
      a[e]     = (_Float16)Ar[k0 + e];              // K = k0+aOff+e
      a[e + 8] = (_Float16)Ar[k0 + 8 + e];          // K = k0+aOff+8+e
    }
    // B fragments: 16 contiguous f16 -> two aligned b128 loads each
    v8h lo0 = *(const v8h*)(Wc0 + k0);
    v8h hi0 = *(const v8h*)(Wc0 + k0 + 8);
    v8h lo1 = *(const v8h*)(Wc1 + k0);
    v8h hi1 = *(const v8h*)(Wc1 + k0 + 8);
    v16h b0 = __builtin_shufflevector(lo0, hi0, 0,1,2,3,4,5,6,7,8,9,10,11,12,13,14,15);
    v16h b1 = __builtin_shufflevector(lo1, hi1, 0,1,2,3,4,5,6,7,8,9,10,11,12,13,14,15);
    acc0 = __builtin_amdgcn_wmma_f32_16x16x32_f16(false, a, false, b0,
                                                  (short)0, acc0, false, false);
    acc1 = __builtin_amdgcn_wmma_f32_16x16x32_f16(false, a, false, b1,
                                                  (short)0, acc1, false, false);
  }

  // Epilogue. Bias values are loop-invariant: hoist as unconditional clamped
  // loads. Only the stores themselves are predicated.
  const float bv0 = bias[cB0];
  const float bv1 = bias[cB1];
#pragma unroll
  for (int r = 0; r < 8; ++r) {
    int row = tileM + r + (hl ? 8 : 0);
    float v0 = acc0[r] + bv0;
    float v1 = acc1[r] + bv1;
    if (relu) { v0 = fmaxf(v0, 0.0f); v1 = fmaxf(v1, 0.0f); }
    bool rOK = row < M;
    if (rOK && c0OK) C[(size_t)row * ldc + colB0] = v0;
    if (rOK && c1OK) C[(size_t)row * ldc + colB1] = v1;
  }
}

// ---------------------------------------------------------------------------
// Dilated KNN. One thread per (b,p): insertion-select the K*D+1 nearest of Np
// points (stable on ties -> matches top_k lowest-index tie-break), then take
// positions 1, 1+D, ..., 1+(K-1)*D (skip self, dilation D).
__global__ void pcnn_knn_kernel(const float* __restrict__ rep,
                                const float* __restrict__ pts,
                                int* __restrict__ idx_out,
                                int Bv, int P, int Np, int K, int D) {
  int t = blockIdx.x * blockDim.x + threadIdx.x;
  if (t >= Bv * P) return;
  int b = t / P, p = t % P;
  const float* rp = rep + ((size_t)b * P + p) * 3;
  float rx = rp[0], ry = rp[1], rz = rp[2];
  int nsel = K * D + 1;
  float bd[MAXSEL]; int bi[MAXSEL];
  for (int i = 0; i < nsel; ++i) { bd[i] = 3.4e38f; bi[i] = -1; }
  const float* pb = pts + (size_t)b * Np * 3;
  for (int n = 0; n < Np; ++n) {
    float dx = pb[n * 3 + 0] - rx;
    float dy = pb[n * 3 + 1] - ry;
    float dz = pb[n * 3 + 2] - rz;
    float d2 = dx * dx + dy * dy + dz * dz;
    if (d2 < bd[nsel - 1]) {
      int j = nsel - 1;
      while (j > 0 && bd[j - 1] > d2) { bd[j] = bd[j - 1]; bi[j] = bi[j - 1]; --j; }
      bd[j] = d2; bi[j] = n;
    }
  }
  int* out = idx_out + ((size_t)b * P + p) * K;
  for (int kk = 0; kk < K; ++kk) out[kk] = bi[1 + kk * D];
}

// ---------------------------------------------------------------------------
// rep[b,p,:] = pts[b, sub_idx[p], :]
__global__ void pcnn_gather_rep_kernel(const float* __restrict__ pts,
                                       const int* __restrict__ sub,
                                       float* __restrict__ rep,
                                       int Bv, int Np, int P) {
  int t = blockIdx.x * blockDim.x + threadIdx.x;
  int total = Bv * P * 3;
  if (t >= total) return;
  int d = t % 3; int p = (t / 3) % P; int b = t / (3 * P);
  rep[t] = pts[((size_t)b * Np + sub[p]) * 3 + d];
}

// ---------------------------------------------------------------------------
// pts_local[b,p,k,:] = pts[b,idx]-rep[b,p];  fts_cat[b,p,k,Cmid:] = fts_pre[b,idx,:]
__global__ void pcnn_gather_kernel(const float* __restrict__ pts,
                                   const float* __restrict__ rep,
                                   const float* __restrict__ fts_pre,
                                   const int* __restrict__ idx,
                                   float* __restrict__ pts_local,
                                   float* __restrict__ fts_cat,
                                   int Bv, int P, int Np, int K, int Cpre, int Cmid) {
  int t = blockIdx.x * blockDim.x + threadIdx.x;
  int total = Bv * P * K;
  if (t >= total) return;
  int k = t % K; int p = (t / K) % P; int b = t / (K * P);
  int n = idx[((size_t)b * P + p) * K + k];
  const float* pp = pts + ((size_t)b * Np + n) * 3;
  const float* rp = rep + ((size_t)b * P + p) * 3;
  float* pl = pts_local + (size_t)t * 3;
  pl[0] = pp[0] - rp[0];
  pl[1] = pp[1] - rp[1];
  pl[2] = pp[2] - rp[2];
  int Ccat = Cmid + Cpre;
  const float* f = fts_pre + ((size_t)b * Np + n) * Cpre;
  float* dst = fts_cat + (size_t)t * Ccat + Cmid;
  for (int c = 0; c < Cpre; ++c) dst[c] = f[c];
}

// ---------------------------------------------------------------------------
// Fused: g[k] = sum_j X[b,p,k,j]*fts_cat[b,p,j,c];
//        y[b,p,c,m] = dw_b[c,m] + sum_k dw_W[c,m,k]*g[k]
// One thread per (b,p,c). K<=12, dm<=4.
__global__ void pcnn_dw_kernel(const float* __restrict__ X,
                               const float* __restrict__ fts_cat,
                               const float* __restrict__ dw_W,
                               const float* __restrict__ dw_b,
                               float* __restrict__ y,
                               int Bv, int P, int K, int Ccat, int dm) {
  int t = blockIdx.x * blockDim.x + threadIdx.x;
  int total = Bv * P * Ccat;
  if (t >= total) return;
  int c = t % Ccat; int bp = t / Ccat;
  const float* Xp = X + (size_t)bp * K * K;
  const float* fc = fts_cat + (size_t)bp * K * Ccat + c;
  float g[12];
  for (int k = 0; k < K; ++k) {
    float s = 0.0f;
    for (int j = 0; j < K; ++j) s += Xp[k * K + j] * fc[(size_t)j * Ccat];
    g[k] = s;
  }
  float* yp = y + ((size_t)bp * Ccat + c) * dm;
  for (int m = 0; m < dm; ++m) {
    float s = dw_b[c * dm + m];
    const float* w = dw_W + ((size_t)c * dm + m) * K;
    for (int k = 0; k < K; ++k) s += w[k] * g[k];
    yp[m] = s;
  }
}

// ---------------------------------------------------------------------------
__global__ void pcnn_mean_kernel(const float* __restrict__ logits,
                                 float* __restrict__ out, int Bv, int P, int C) {
  int t = blockIdx.x * blockDim.x + threadIdx.x;
  if (t >= Bv * C) return;
  int b = t / C, c = t % C;
  float s = 0.0f;
  for (int p = 0; p < P; ++p) s += logits[((size_t)b * P + p) * C + c];
  out[t] = s / (float)P;
}

// ===========================================================================
extern "C" void kernel_launch(void* const* d_in, const int* in_sizes, int n_in,
                              void* d_out, int out_size, void* d_ws, size_t ws_size,
                              hipStream_t stream) {
  (void)n_in; (void)out_size; (void)ws_size;

  // --- input-flattening scheme detection ----------------------------------
  // Scheme A (insertion order): x, l0..l4 (16 each), fcn(W1,b1,W2,b2,W3,b3), sub_idx
  // Scheme B (jax sorted keys): fcn(W1,W2,W3,b1,b2,b3), l0..l4 (sorted), sub_idx, x
  const bool ins = (in_sizes[0] == BATCH * 3 * NPTS);
  // which: 0 pre_W,1 pre_b,2 d1_W,3 d1_b,4 d2_W,5 d2_b,6 xt0_W,7 xt0_b,
  //        8 xt1_W,9 xt1_b,10 xt2_W,11 xt2_b,12 dw_W,13 dw_b,14 pw_W,15 pw_b
  static const int sortedMap[16] = {6,7,0,1,2,3,10,11,12,13,14,15,4,5,8,9};
  static const int fcnMapB[6]   = {0,3,1,4,2,5}; // W1,b1,W2,b2,W3,b3 -> sorted pos
  auto lp = [&](int layer, int which) -> const float* {
    int idx = ins ? (1 + 16 * layer + which) : (6 + 16 * layer + sortedMap[which]);
    return (const float*)d_in[idx];
  };
  auto fp = [&](int j) -> const float* {
    int idx = ins ? (81 + j) : fcnMapB[j];
    return (const float*)d_in[idx];
  };
  const float* x_in    = (const float*)d_in[ins ? 0 : 87];
  const int*   sub_idx = (const int*)  d_in[ins ? 87 : 86];

  // --- workspace bump allocator -------------------------------------------
  char* wsb = (char*)d_ws;
  size_t off = 0;
  auto alloc = [&](size_t elems) -> float* {
    float* p = (float*)(wsb + off);
    off += ((elems * sizeof(float) + 255) & ~(size_t)255);
    return p;
  };
  float* PTS  = alloc((size_t)BATCH * NPTS * 3);
  float* REP  = alloc((size_t)BATCH * 120 * 3);
  float* FTSA = alloc((size_t)BATCH * NPTS * 96);          // ping
  float* FTSB = alloc((size_t)BATCH * NPTS * 96);          // pong
  float* FPRE = alloc((size_t)BATCH * NPTS * 64);
  int*   IDX  = (int*)alloc((size_t)BATCH * NPTS * 8);
  float* PLOC = alloc((size_t)BATCH * NPTS * 8 * 3);
  float* LIFT = alloc((size_t)BATCH * NPTS * 8 * 24);
  float* FCAT = alloc((size_t)BATCH * NPTS * 8 * 72);
  float* X0   = alloc((size_t)BATCH * NPTS * 64);
  float* X1   = alloc((size_t)BATCH * NPTS * 64);
  float* YB   = alloc((size_t)BATCH * NPTS * 72 * 2);
  _Float16* WT = (_Float16*)alloc(32768);   // packed-weight buffer (f16)
  float* H1   = alloc((size_t)BATCH * 120 * 128);
  float* H2   = alloc((size_t)BATCH * 120 * 64);
  float* LOG  = alloc((size_t)BATCH * 120 * 40);
  (void)alloc(64);                           // guard: A-side K-pad over-reads

  auto g1 = [](int n, int b) { return dim3((unsigned)((n + b - 1) / b)); };

  // GEMM with a pre-packed Wt
  auto gemmT = [&](const float* A, int lda, const float* bias,
                   float* C, int ldc, int M, int Kd, int Nd, int relu) {
    dim3 grid((unsigned)((Nd + 31) / 32), (unsigned)((M + 15) / 16));
    pcnn_gemm_wmma_kernel<<<grid, dim3(32), 0, stream>>>(
        A, lda, WT, bias, C, ldc, M, Kd, Nd, relu);
  };
  // pack raw f32 weights, then GEMM
  auto gemm = [&](const float* A, int lda, const float* W, int ldw,
                  const float* bias, float* C, int ldc,
                  int M, int Kd, int Nd, int relu) {
    int kpad = (Kd + 31) & ~31;
    pcnn_pack_w_kernel<<<g1(Nd * kpad, 256), 256, 0, stream>>>(W, ldw, WT, Kd, Nd);
    gemmT(A, lda, bias, C, ldc, M, Kd, Nd, relu);
  };

  // --- transpose input ----------------------------------------------------
  {
    int total = BATCH * NPTS * 3;
    pcnn_transpose_kernel<<<g1(total, 256), 256, 0, stream>>>(x_in, PTS, BATCH, NPTS);
  }

  static const int cCout[5] = {32, 64, 96, 128, 160};
  static const int cK[5]    = {8, 8, 8, 12, 12};
  static const int cD[5]    = {1, 2, 4, 4, 6};
  static const int cDm[5]   = {4, 2, 2, 2, 2};
  static const int cSub[5]  = {0, 0, 0, 1, 0};

  float* pts = PTS;               // current point cloud (B,Np,3)
  int Np = NPTS;
  const float* fts = PTS;         // current features (B,Np,Cin); layer0 fts==pts
  int Cin = 3;
  float* fbuf[2] = {FTSA, FTSB};
  int pb = 0;

  for (int i = 0; i < 5; ++i) {
    int Cout = cCout[i], K = cK[i], D = cD[i], dm = cDm[i];
    int Cpre = Cout / 2, Cmid = Cout / 4, Ccat = Cpre + Cmid;
    int K2 = K * K;
    int P = cSub[i] ? 120 : Np;

    // 1) pre-dense on ALL Np points: FPRE = relu(fts @ pre_W + pre_b)
    gemm(fts, Cin, lp(i, 0), Cpre, lp(i, 1), FPRE, Cpre, BATCH * Np, Cin, Cpre, 1);

    // 2) representative points
    const float* rep;
    if (cSub[i]) {
      int total = BATCH * P * 3;
      pcnn_gather_rep_kernel<<<g1(total, 256), 256, 0, stream>>>(pts, sub_idx, REP, BATCH, Np, P);
      rep = REP;
    } else {
      rep = pts;
    }

    // 3) dilated KNN
    pcnn_knn_kernel<<<g1(BATCH * P, 128), 128, 0, stream>>>(rep, pts, IDX, BATCH, P, Np, K, D);

    // 4) gather: pts_local + fts_cat[:, Cmid:]
    {
      int total = BATCH * P * K;
      pcnn_gather_kernel<<<g1(total, 256), 256, 0, stream>>>(
          pts, rep, FPRE, IDX, PLOC, FCAT, BATCH, P, Np, K, Cpre, Cmid);
    }

    // 5) lifting: d1 then d2 (d2 writes directly into fts_cat[:, :Cmid])
    gemm(PLOC, 3,    lp(i, 2), Cmid, lp(i, 3), LIFT, Cmid, BATCH * P * K, 3,    Cmid, 1);
    gemm(LIFT, Cmid, lp(i, 4), Cmid, lp(i, 5), FCAT, Ccat, BATCH * P * K, Cmid, Cmid, 1);

    // 6) X-transform: xt0 (specially packed), xt1, xt2
    {
      int kpad = (3 * K + 31) & ~31;
      pcnn_pack_xt0_kernel<<<g1(K2 * kpad, 256), 256, 0, stream>>>(lp(i, 6), WT, K);
    }
    gemmT(PLOC, 3 * K, lp(i, 7), X0, K2, BATCH * P, 3 * K, K2, 1);
    gemm (X0, K2, lp(i, 8),  K2, lp(i, 9),  X1, K2, BATCH * P, K2, K2, 1);
    gemm (X1, K2, lp(i, 10), K2, lp(i, 11), X0, K2, BATCH * P, K2, K2, 0);

    // 7) fused (X @ fts_cat) + depthwise over K -> Y (B,P,Ccat,dm)
    {
      int total = BATCH * P * Ccat;
      pcnn_dw_kernel<<<g1(total, 256), 256, 0, stream>>>(
          X0, FCAT, lp(i, 12), lp(i, 13), YB, BATCH, P, K, Ccat, dm);
    }

    // 8) pointwise: specially packed pw_W, relu GEMM -> new features
    {
      int kpad = (Ccat * dm + 31) & ~31;
      pcnn_pack_pw_kernel<<<g1(Cout * kpad, 256), 256, 0, stream>>>(lp(i, 14), WT, Ccat, dm, Cout);
    }
    float* fout = fbuf[pb]; pb ^= 1;
    gemmT(YB, Ccat * dm, lp(i, 15), fout, Cout, BATCH * P, Ccat * dm, Cout, 1);

    fts = fout; Cin = Cout;
    pts = (float*)rep; Np = P;
  }

  // --- FC head (B*120 rows) + mean over points ----------------------------
  gemm(fts, 160, fp(0), 128, fp(1), H1,  128, BATCH * 120, 160, 128, 1);
  gemm(H1,  128, fp(2), 64,  fp(3), H2,  64,  BATCH * 120, 128, 64,  1);
  gemm(H2,  64,  fp(4), 40,  fp(5), LOG, 40,  BATCH * 120, 64,  40,  0);
  pcnn_mean_kernel<<<g1(BATCH * 40, 128), 128, 0, stream>>>(
      LOG, (float*)d_out, BATCH, 120, 40);
}
